// HV_OCTAMambaBlock_48782238548075
// MI455X (gfx1250) — matrix operand order repaired and checked by reading, not compiled, not run on hardware
//
#include <hip/hip_runtime.h>
#include <hip/hip_bf16.h>
#include <math.h>

typedef _Float16 f16;
typedef _Float16 v16h __attribute__((ext_vector_type(16)));
typedef float    v8f  __attribute__((ext_vector_type(8)));
typedef float    v4f  __attribute__((ext_vector_type(4)));
typedef unsigned int v4u __attribute__((ext_vector_type(4)));
typedef int      v4i  __attribute__((ext_vector_type(4)));
typedef int      v8i  __attribute__((ext_vector_type(8)));

#define LPIX 4096
#define HWD  64

#if defined(__has_builtin)
#if __has_builtin(__builtin_amdgcn_tensor_load_to_lds) && __has_builtin(__builtin_amdgcn_s_wait_tensorcnt)
#define HAVE_TDM 1
#endif
#endif
#ifndef HAVE_TDM
#define HAVE_TDM 0
#endif

__device__ __forceinline__ float gelu_f(float x) {
    return 0.5f * x * (1.0f + erff(x * 0.70710678118654752f));
}
__device__ __forceinline__ float silu_f(float x) {
    return x / (1.0f + expf(-x));
}

#if HAVE_TDM
// Issue a TDM load of a 32(row) x 64(col) f32 tile (row stride LPIX) to LDS.
// D# layout per CDNA5 ISA 8.3-8.5: group0 {count=1, lds_addr, global_addr, type=2},
// group1 {data_size=4B, tensor_dim0=64, tensor_dim1=32, tile_dim0=64, tile_dim1=32,
// tensor_dim0_stride=4096}; groups 2/3 zero (2-D tensor).
__device__ __forceinline__ void tdm_load_slab(const float* gptr, unsigned lds_off)
{
    const unsigned long long ga = (unsigned long long)gptr;
    v4u g0;
    g0[0] = 0x1u;                                   // count=1, user descriptor
    g0[1] = lds_off;                                // LDS byte address
    g0[2] = (unsigned)(ga & 0xFFFFFFFFu);           // global_addr[31:0]
    g0[3] = (unsigned)((ga >> 32) & 0x1FFFFFFu) | 0x80000000u;  // ga[56:32] | type=2
    v8i g1;
    g1[0] = 0x00020000;                             // data_size=2 (4B), mask=0
    g1[1] = (int)(64u << 16);                       // tensor_dim0 = 64 (low 16)
    g1[2] = (int)(32u << 16);                       // tensor_dim1 = 32 (low 16)
    g1[3] = (int)(64u << 16);                       // tile_dim0 = 64
    g1[4] = 32;                                     // tile_dim1 = 32, tile_dim2 = 0
    g1[5] = 4096;                                   // tensor_dim0_stride[31:0]
    g1[6] = 0;                                      // stride0 hi / stride1 lo
    g1[7] = 0;                                      // stride1 hi
    v4i gz; gz[0] = 0; gz[1] = 0; gz[2] = 0; gz[3] = 0;
#if defined(__clang_major__) && (__clang_major__ >= 23)
    v8i gz8; gz8[0]=0; gz8[1]=0; gz8[2]=0; gz8[3]=0; gz8[4]=0; gz8[5]=0; gz8[6]=0; gz8[7]=0;
    __builtin_amdgcn_tensor_load_to_lds(g0, g1, gz, gz, gz8, 0);
#else
    __builtin_amdgcn_tensor_load_to_lds(g0, g1, gz, gz, 0);
#endif
}
#endif

__device__ __forceinline__ v16h load_a_frag(const float* pA)
{
    const v4f a0 = *(const v4f*)(pA);
    const v4f a1 = *(const v4f*)(pA + 4);
    const v4f a2 = *(const v4f*)(pA + 8);
    const v4f a3 = *(const v4f*)(pA + 12);
    v16h av;
#pragma unroll
    for (int j = 0; j < 4; ++j) {
        av[j]      = (f16)a0[j];
        av[4 + j]  = (f16)a1[j];
        av[8 + j]  = (f16)a2[j];
        av[12 + j] = (f16)a3[j];
    }
    return av;
}

// ---------------------------------------------------------------------------
// Implicit-GEMM conv via WMMA.  Y[b,m,p] = ACT(alpha*sum_k W[m,k]*X[b,c(k),shift_t(p)]+bias)*mul
// k = c*T + t ; tap t shifts pixel by (dx0+t*ddx, dy0+t*ddy), zero pad (SAME).
// One wave per 16(M) x 64(N) strip (strip = one image row).
// T==1 main loop: B slab (32K x 64N) staged to LDS by the Tensor Data Mover,
// double buffered (tensor_load_to_lds + s_wait_tensorcnt); A via b128 loads.
// f16 inputs, f32 accumulate; 4 WMMA per A fragment.
// ---------------------------------------------------------------------------
template<int ACT, int T>
__global__ __launch_bounds__(32)
void wmma_gemm(const float* __restrict__ Wm, const float* __restrict__ Xm,
               const float* __restrict__ bias, const float* __restrict__ mul,
               float* __restrict__ Ym,
               int O, int C, int dx0, int dy0, int ddx, int ddy,
               long Xbs, long Ybs, long Mulbs, float alpha)
{
    const int lane = threadIdx.x;
    const int half = lane >> 4;
    const int lidx = lane & 15;
    const int n0 = blockIdx.x * 64;          // 64-wide N strip == image row py
    const int m0 = blockIdx.y * 16;
    const int b  = blockIdx.z;
    const int py = blockIdx.x;

    const float* X = Xm + (long)b * Xbs;
    float*       Y = Ym + (long)b * Ybs;

    const int K     = C * T;
    const int mrow  = m0 + lidx;
    const int mclmp = (mrow < O) ? mrow : (O - 1);
    const float* Wrow = Wm + (long)mclmp * K;
    const float* pA = Wrow + (half << 4);

    v8f acc[4] = {};
    const int kmain = K & ~31;

    if (T == 1) {
#if HAVE_TDM
        __shared__ float ldsb[2 * 2048];               // 2 x (32x64) f32 slabs
        const int nslab = kmain >> 5;
        const float* gbase = X + n0;
        if (nslab > 0) tdm_load_slab(gbase, 0u);
        for (int s = 0; s < nslab; ++s) {
            if (s + 1 < nslab)
                tdm_load_slab(gbase + ((long)(s + 1) << 5) * LPIX,
                              (unsigned)(((s + 1) & 1) << 13));
            const v16h av = load_a_frag(pA);
            pA += 32;
            if (s + 1 < nslab) __builtin_amdgcn_s_wait_tensorcnt(1);
            else               __builtin_amdgcn_s_wait_tensorcnt(0);
            const float* lb = &ldsb[((s & 1) << 11) + (half << 10) + lidx];
#pragma unroll
            for (int nt = 0; nt < 4; ++nt) {
                float tmp[16];
#pragma unroll
                for (int i = 0; i < 16; ++i)
                    tmp[i] = lb[i * 64 + nt * 16];
                v16h bv;
#pragma unroll
                for (int i = 0; i < 16; ++i)
                    bv[i] = (f16)tmp[i];
                acc[nt] = __builtin_amdgcn_wmma_f32_16x16x32_f16(false, av, false, bv,
                                                                 (short)0, acc[nt], false, false);
            }
        }
#else
        const float* pB = X + ((long)(half << 4)) * LPIX + py * HWD + lidx;
        for (int k0 = 0; k0 < kmain; k0 += 32) {
            const v16h av = load_a_frag(pA);
            pA += 32;
#pragma unroll
            for (int nt = 0; nt < 4; ++nt) {
                float tmp[16];
#pragma unroll
                for (int i = 0; i < 16; ++i)
                    tmp[i] = pB[(long)i * LPIX + nt * 16];
                v16h bv;
#pragma unroll
                for (int i = 0; i < 16; ++i)
                    bv[i] = (f16)tmp[i];
                acc[nt] = __builtin_amdgcn_wmma_f32_16x16x32_f16(false, av, false, bv,
                                                                 (short)0, acc[nt], false, false);
            }
            pB += 32L * LPIX;
        }
#endif
    } else {
        for (int k0 = 0; k0 < kmain; k0 += 32) {
            const v16h av = load_a_frag(pA);
            pA += 32;
#pragma unroll
            for (int nt = 0; nt < 4; ++nt) {
                const int px = nt * 16 + lidx;
                float tmp[16];
                bool  okm[16];
#pragma unroll
                for (int i = 0; i < 16; ++i) {
                    const int kb = k0 + (half << 4) + i;
                    const int c  = kb / 3;
                    const int t  = kb - 3 * c;
                    const int qx = px + dx0 + t * ddx;
                    const int qy = py + dy0 + t * ddy;
                    okm[i] = ((unsigned)qx < 64u) && ((unsigned)qy < 64u);
                    const int qxc = (qx < 0) ? 0 : ((qx > 63) ? 63 : qx);
                    const int qyc = (qy < 0) ? 0 : ((qy > 63) ? 63 : qy);
                    tmp[i] = X[(long)c * LPIX + qyc * HWD + qxc];
                }
                v16h bv;
#pragma unroll
                for (int i = 0; i < 16; ++i)
                    bv[i] = okm[i] ? (f16)tmp[i] : (f16)0.f;
                acc[nt] = __builtin_amdgcn_wmma_f32_16x16x32_f16(false, av, false, bv,
                                                                 (short)0, acc[nt], false, false);
            }
        }
    }

    // ---- K tail (runs at most once; only T==1 shapes have K%32 != 0)
    if (kmain < K) {
        const int base0 = kmain + (half << 4);
        v16h av;
        {
            float ta[16];
#pragma unroll
            for (int i = 0; i < 16; ++i) {
                const int ka = base0 + i;
                ta[i] = Wrow[(ka < K) ? ka : (K - 1)];
            }
#pragma unroll
            for (int i = 0; i < 16; ++i) {
                const int ka = base0 + i;
                av[i] = (ka < K) ? (f16)ta[i] : (f16)0.f;
            }
        }
#pragma unroll
        for (int nt = 0; nt < 4; ++nt) {
            const int px = nt * 16 + lidx;
            float tmp[16];
            bool  okm[16];
#pragma unroll
            for (int i = 0; i < 16; ++i) {
                const int kb  = kmain + (half << 4) + i;
                const bool kok = kb < K;
                const int kc  = kok ? kb : (K - 1);
                int c, t;
                if (T == 1) { c = kc; t = 0; }
                else        { c = kc / 3; t = kc - 3 * c; }
                const int qx = px + dx0 + t * ddx;
                const int qy = py + dy0 + t * ddy;
                okm[i] = kok && ((unsigned)qx < 64u) && ((unsigned)qy < 64u);
                const int qxc = (qx < 0) ? 0 : ((qx > 63) ? 63 : qx);
                const int qyc = (qy < 0) ? 0 : ((qy > 63) ? 63 : qy);
                tmp[i] = X[(long)c * LPIX + qyc * HWD + qxc];
            }
            v16h bv;
#pragma unroll
            for (int i = 0; i < 16; ++i)
                bv[i] = okm[i] ? (f16)tmp[i] : (f16)0.f;
            acc[nt] = __builtin_amdgcn_wmma_f32_16x16x32_f16(false, av, false, bv,
                                                             (short)0, acc[nt], false, false);
        }
    }

#pragma unroll
    for (int nt = 0; nt < 4; ++nt) {
        const int ncol = n0 + nt * 16 + lidx;
#pragma unroll
        for (int r = 0; r < 8; ++r) {
            const int m = m0 + r + (half << 3);
            if (m < O) {
                float v = acc[nt][r] * alpha;
                if (bias) v += bias[m];
                if (ACT == 1) v = gelu_f(v);
                if (mul) v *= mul[(long)b * Mulbs + (long)m * LPIX + ncol];
                Y[(long)m * LPIX + ncol] = v;
            }
        }
    }
}

// ---------------------------------------------------------------------------
// depthwise 3x3 SAME conv + bias + SiLU on xz rows [0,336): branchless taps
// ---------------------------------------------------------------------------
__global__ void dwconv_silu(const float* __restrict__ xz, const float* __restrict__ w,
                            const float* __restrict__ bias, float* __restrict__ out)
{
    const int p  = blockIdx.x * 256 + threadIdx.x;
    const int bc = blockIdx.y;
    const int b = bc / 336, c = bc % 336;
    if (p >= LPIX) return;
    const float* src = xz + ((long)b * 672 + c) * LPIX;
    const int py = p >> 6, px = p & 63;
    float tmp[9];
    bool  okm[9];
#pragma unroll
    for (int ky = 0; ky < 3; ++ky)
#pragma unroll
        for (int kx = 0; kx < 3; ++kx) {
            const int qy = py + ky - 1, qx = px + kx - 1;
            const int qyc = (qy < 0) ? 0 : ((qy > 63) ? 63 : qy);
            const int qxc = (qx < 0) ? 0 : ((qx > 63) ? 63 : qx);
            okm[ky * 3 + kx] = ((unsigned)qy < 64u) && ((unsigned)qx < 64u);
            tmp[ky * 3 + kx] = src[qyc * HWD + qxc];
        }
    float acc = bias[c];
#pragma unroll
    for (int j = 0; j < 9; ++j)
        acc += okm[j] ? (w[c * 9 + j] * tmp[j]) : 0.f;
    out[((long)b * 336 + c) * LPIX + p] = silu_f(acc);
}

// ---------------------------------------------------------------------------
// Selective scan: one thread per (b,k,c); direction k reads/writes through its
// pixel permutation.  All 44 per-step operands gathered off one base pointer
// (immediate offsets) before use -> single load clause per step.
// ---------------------------------------------------------------------------
__global__ void scan_kernel(const float* __restrict__ xi, const float* __restrict__ Z,
                            const float* __restrict__ dtw, const float* __restrict__ dtb,
                            const float* __restrict__ Alog, const float* __restrict__ Dp,
                            float* __restrict__ oy)
{
    const int idx = blockIdx.x * blockDim.x + threadIdx.x;
    if (idx >= 4 * 4 * 336) return;
    const int c = idx % 336;
    const int k = (idx / 336) % 4;
    const int b = idx / (4 * 336);

    const float* u_row = xi + ((long)b * 336 + c) * LPIX;
    const float* Zk    = Z  + ((long)(b * 4 + k) * 43) * LPIX;
    float*       orow  = oy + ((long)((b * 4 + k) * 336) + c) * LPIX;

    float A[16], h[16], wdt[11];
#pragma unroll
    for (int n = 0; n < 16; ++n) {
        A[n] = -expf(Alog[((long)k * 336 + c) * 16 + n]);
        h[n] = 0.f;
    }
#pragma unroll
    for (int r = 0; r < 11; ++r) wdt[r] = dtw[((long)k * 336 + c) * 11 + r];
    const float bdt = dtb[k * 336 + c];
    const float Dv  = Dp[k * 336 + c];

    for (int l = 0; l < LPIX; ++l) {
        int p;
        if (k == 0)      p = l;
        else if (k == 1) p = ((l & 63) << 6) | (l >> 6);
        else if (k == 2) p = 4095 - l;
        else { const int q = 4095 - l; p = ((q & 63) << 6) | (q >> 6); }

        if (l + 1 < LPIX) {   // prefetch next step (global_prefetch_b8)
            int l2 = l + 1, p2;
            if (k == 0)      p2 = l2;
            else if (k == 1) p2 = ((l2 & 63) << 6) | (l2 >> 6);
            else if (k == 2) p2 = 4095 - l2;
            else { const int q2 = 4095 - l2; p2 = ((q2 & 63) << 6) | (q2 >> 6); }
            __builtin_prefetch(u_row + p2, 0, 1);
            __builtin_prefetch(Zk + p2, 0, 1);
        }

        // gather all operands first (one base, immediate offsets)
        const float* zp = Zk + p;
        float zd_[11], zb_[16], zc_[16];
#pragma unroll
        for (int r = 0; r < 11; ++r) zd_[r] = zp[(long)r * LPIX];
#pragma unroll
        for (int n = 0; n < 16; ++n) zb_[n] = zp[(long)(11 + n) * LPIX];
#pragma unroll
        for (int n = 0; n < 16; ++n) zc_[n] = zp[(long)(27 + n) * LPIX];
        const float u = u_row[p];

        float s = bdt;
#pragma unroll
        for (int r = 0; r < 11; ++r) s += wdt[r] * zd_[r];
        const float dlt = (s > 30.f) ? s : log1pf(expf(s));
        const float du  = dlt * u;
        float yv = 0.f;
#pragma unroll
        for (int n = 0; n < 16; ++n) {
            h[n] = h[n] * expf(dlt * A[n]) + du * zb_[n];
            yv  += h[n] * zc_[n];
        }
        orow[p] = yv + Dv * u;
    }
}

__global__ void combine_kernel(const float* __restrict__ oy, float* __restrict__ yc)
{
    const long j = (long)blockIdx.x * blockDim.x + threadIdx.x;
    const long per_b = 336L * LPIX;
    if (j >= 4 * per_b) return;
    const long b = j / per_b, r = j % per_b;
    const float* base = oy + b * 4 * per_b;
    yc[j] = base[r] + base[per_b + r] + base[2 * per_b + r] + base[3 * per_b + r];
}

__global__ void ln_gate_kernel(const float* __restrict__ yc, const float* __restrict__ g,
                               const float* __restrict__ bt, const float* __restrict__ xz,
                               float* __restrict__ y2)
{
    const int t = blockIdx.x * blockDim.x + threadIdx.x;
    if (t >= 4 * LPIX) return;
    const int b = t / LPIX, p = t % LPIX;
    const float* base = yc + (long)b * 336 * LPIX + p;
    float m = 0.f;
    for (int c = 0; c < 336; c += 4) {
        const float t0 = base[(long)(c + 0) * LPIX], t1 = base[(long)(c + 1) * LPIX];
        const float t2 = base[(long)(c + 2) * LPIX], t3 = base[(long)(c + 3) * LPIX];
        m += (t0 + t1) + (t2 + t3);
    }
    m *= (1.0f / 336.f);
    float v = 0.f;
    for (int c = 0; c < 336; c += 4) {
        const float t0 = base[(long)(c + 0) * LPIX] - m, t1 = base[(long)(c + 1) * LPIX] - m;
        const float t2 = base[(long)(c + 2) * LPIX] - m, t3 = base[(long)(c + 3) * LPIX] - m;
        v += (t0 * t0 + t1 * t1) + (t2 * t2 + t3 * t3);
    }
    v *= (1.0f / 336.f);
    const float rs = rsqrtf(v + 1e-5f);
    const float* zb = xz + ((long)b * 672 + 336) * LPIX + p;
    float* ob = y2 + (long)b * 336 * LPIX + p;
    for (int c = 0; c < 336; ++c) {
        const float val = (base[(long)c * LPIX] - m) * rs * g[c] + bt[c];
        ob[(long)c * LPIX] = val * silu_f(zb[(long)c * LPIX]);
    }
}

__global__ void ln96_kernel(const float* __restrict__ X, const float* __restrict__ g,
                            const float* __restrict__ bt, float* __restrict__ Y, float eps)
{
    const int t = blockIdx.x * blockDim.x + threadIdx.x;
    if (t >= 4 * LPIX) return;
    const int b = t / LPIX, p = t % LPIX;
    const float* base = X + (long)b * 96 * LPIX + p;
    float m = 0.f;
    for (int c = 0; c < 96; c += 4) {
        const float t0 = base[(c + 0) * LPIX], t1 = base[(c + 1) * LPIX];
        const float t2 = base[(c + 2) * LPIX], t3 = base[(c + 3) * LPIX];
        m += (t0 + t1) + (t2 + t3);
    }
    m *= (1.0f / 96.f);
    float v = 0.f;
    for (int c = 0; c < 96; c += 4) {
        const float t0 = base[(c + 0) * LPIX] - m, t1 = base[(c + 1) * LPIX] - m;
        const float t2 = base[(c + 2) * LPIX] - m, t3 = base[(c + 3) * LPIX] - m;
        v += (t0 * t0 + t1 * t1) + (t2 * t2 + t3 * t3);
    }
    v *= (1.0f / 96.f);
    const float rs = rsqrtf(v + eps);
    float* ob = Y + (long)b * 96 * LPIX + p;
    for (int c = 0; c < 96; ++c)
        ob[c * LPIX] = (base[c * LPIX] - m) * rs * g[c] + bt[c];
}

__global__ void final_kernel(const float* __restrict__ X, const float* __restrict__ g,
                             const float* __restrict__ bt, const float* __restrict__ skip,
                             const float* __restrict__ scale, float* __restrict__ Y)
{
    const int t = blockIdx.x * blockDim.x + threadIdx.x;
    if (t >= 4 * LPIX) return;
    const int b = t / LPIX, p = t % LPIX;
    const float* base = X + (long)b * 96 * LPIX + p;
    float m = 0.f;
    for (int c = 0; c < 96; c += 4) {
        const float t0 = base[(c + 0) * LPIX], t1 = base[(c + 1) * LPIX];
        const float t2 = base[(c + 2) * LPIX], t3 = base[(c + 3) * LPIX];
        m += (t0 + t1) + (t2 + t3);
    }
    m *= (1.0f / 96.f);
    float v = 0.f;
    for (int c = 0; c < 96; c += 4) {
        const float t0 = base[(c + 0) * LPIX] - m, t1 = base[(c + 1) * LPIX] - m;
        const float t2 = base[(c + 2) * LPIX] - m, t3 = base[(c + 3) * LPIX] - m;
        v += (t0 * t0 + t1 * t1) + (t2 * t2 + t3 * t3);
    }
    v *= (1.0f / 96.f);
    const float rs = rsqrtf(v + 1e-5f);
    const float sc = scale[0];
    const float* sk = skip + (long)b * 96 * LPIX + p;
    float* ob = Y + (long)b * 96 * LPIX + p;
    for (int c = 0; c < 96; ++c) {
        const float val = gelu_f((base[c * LPIX] - m) * rs * g[c] + bt[c]);
        ob[c * LPIX] = val + sk[c * LPIX] * sc;
    }
}

__global__ void mul_kernel(const float* __restrict__ A, long abs_,
                           const float* __restrict__ Bm, long bbs,
                           float* __restrict__ Y, long ybs, int Crows)
{
    const long j = (long)blockIdx.x * blockDim.x + threadIdx.x;
    const long per_b = (long)Crows * LPIX;
    if (j >= 4 * per_b) return;
    const long b = j / per_b, r = j % per_b;
    Y[b * ybs + r] = A[b * abs_ + r] * Bm[b * bbs + r];
}

__global__ void add_kernel(const float* __restrict__ A, const float* __restrict__ B,
                           float* __restrict__ Y, long n)
{
    const long j = (long)blockIdx.x * blockDim.x + threadIdx.x;
    if (j < n) Y[j] = A[j] + B[j];
}

// ---------------------------------------------------------------------------
static inline void gemm(hipStream_t s, int act, int T,
                        const float* W, const float* X, const float* bias,
                        const float* mul, float* Y,
                        int O, int C, int dx0, int dy0, int ddx, int ddy,
                        long Xbs, long Ybs, long Mbs, float alpha)
{
    dim3 grid(64, (O + 15) / 16, 4), blk(32);
    if (T == 1) {
        if (act == 1) wmma_gemm<1, 1><<<grid, blk, 0, s>>>(W, X, bias, mul, Y, O, C, dx0, dy0, ddx, ddy, Xbs, Ybs, Mbs, alpha);
        else          wmma_gemm<0, 1><<<grid, blk, 0, s>>>(W, X, bias, mul, Y, O, C, dx0, dy0, ddx, ddy, Xbs, Ybs, Mbs, alpha);
    } else {
        if (act == 1) wmma_gemm<1, 3><<<grid, blk, 0, s>>>(W, X, bias, mul, Y, O, C, dx0, dy0, ddx, ddy, Xbs, Ybs, Mbs, alpha);
        else          wmma_gemm<0, 3><<<grid, blk, 0, s>>>(W, X, bias, mul, Y, O, C, dx0, dy0, ddx, ddy, Xbs, Ybs, Mbs, alpha);
    }
}

extern "C" void kernel_launch(void* const* d_in, const int* in_sizes, int n_in,
                              void* d_out, int out_size, void* d_ws, size_t ws_size,
                              hipStream_t stream)
{
    const long L = LPIX;
    const float* x          = (const float*)d_in[0];
    const float* res_w      = (const float*)d_in[1];
    const float* res_b      = (const float*)d_in[2];
    const float* scale      = (const float*)d_in[3];
    const float* ln1_w      = (const float*)d_in[4];
    const float* ln1_b      = (const float*)d_in[5];
    const float* ln_w       = (const float*)d_in[6];
    const float* ln_b       = (const float*)d_in[7];
    const float* pw_w       = (const float*)d_in[8];
    const float* pw_b       = (const float*)d_in[9];
    const float* fuse_w     = (const float*)d_in[10];
    const float* fuse_b     = (const float*)d_in[11];
    const float* bh_w[3]    = {(const float*)d_in[12], (const float*)d_in[16], (const float*)d_in[20]};
    const float* bh_b[3]    = {(const float*)d_in[13], (const float*)d_in[17], (const float*)d_in[21]};
    const float* bv_w[3]    = {(const float*)d_in[14], (const float*)d_in[18], (const float*)d_in[22]};
    const float* bv_b[3]    = {(const float*)d_in[15], (const float*)d_in[19], (const float*)d_in[23]};
    const float* proj_in_w  = (const float*)d_in[24];
    const float* proj_in_b  = (const float*)d_in[25];
    const float* pw0_w      = (const float*)d_in[26];
    const float* pw0_b      = (const float*)d_in[27];
    const float* pw1_w      = (const float*)d_in[28];
    const float* pw1_b      = (const float*)d_in[29];
    const float* proj_out_w = (const float*)d_in[30];
    const float* proj_out_b = (const float*)d_in[31];
    const float* in_proj_w  = (const float*)d_in[32];
    const float* conv_w     = (const float*)d_in[33];
    const float* conv_b     = (const float*)d_in[34];
    const float* x_proj_w   = (const float*)d_in[35];
    const float* dt_proj_w  = (const float*)d_in[36];
    const float* dt_proj_b  = (const float*)d_in[37];
    const float* A_log      = (const float*)d_in[38];
    const float* Dp         = (const float*)d_in[39];
    const float* out_nw     = (const float*)d_in[40];
    const float* out_nb     = (const float*)d_in[41];
    const float* out_proj_w = (const float*)d_in[42];

    float* ws    = (float*)d_ws;
    float* bufS  = ws;                   // skip           4*96*L
    float* bufMY = bufS  + 1572864;      // msdac out      4*96*L
    float* bufA  = bufMY + 1572864;      // y0 / yn / attn 4*96*L
    float* bufB  = bufA  + 1572864;      // tmp / y_c / ys 4*96*L
    float* bufF  = bufB  + 1572864;      // fused          4*192*L
    float* bufC  = bufF  + 3145728;      // concat / Z / gn 4*288*L
    float* bufXZ = bufC  + 4718592;      // xz             4*672*L
    float* bufXI = bufXZ + 11010048;     // xi -> y_comb   4*336*L
    float* bufOY = bufXI + 5505024;      // scan planes    4*4*336*L
    float* bufY2 = bufOY;                // reuse: gated y 4*336*L
    float* bufDW = bufOY + 5505024;      // dw = ss2d/3    4*168*L
    float* bufGA = bufC;                 // gn y_a (24ch)
    float* bufGB = bufC + 1048576;       // gn y_b (48ch)

    // 1. residual 1x1 conv
    gemm(stream, 0, 1, res_w, x, res_b, nullptr, bufS, 96, 48, 0,0,0,0, 48L*L, 96L*L, 0, 1.f);
    // 2. MSDAC stem + exact-erf GELU
    gemm(stream, 1, 1, pw_w, x, pw_b, nullptr, bufA, 96, 48, 0,0,0,0, 48L*L, 96L*L, 0, 1.f);
    // 3. three dilated asymmetric branches (implicit-GEMM convs)
    for (int d = 1; d <= 3; ++d) {
        gemm(stream, 0, 3, bh_w[d-1], bufA, bh_b[d-1], nullptr, bufB,
             96, 96, -d, 0, d, 0, 96L*L, 96L*L, 0, 1.f);
        gemm(stream, 1, 3, bv_w[d-1], bufB, bv_b[d-1], nullptr, bufC + (long)(d-1)*96*L,
             96, 96, 0, -d, 0, d, 96L*L, 288L*L, 0, 1.f);
    }
    // 4. fuse 1x1 (288 -> 96)
    gemm(stream, 0, 1, fuse_w, bufC, fuse_b, nullptr, bufMY, 96, 288, 0,0,0,0, 288L*L, 96L*L, 0, 1.f);
    // 5. LN1 (eps 1e-6)
    ln96_kernel<<<(4*LPIX+255)/256, 256, 0, stream>>>(bufMY, ln1_w, ln1_b, bufA, 1e-6f);
    // 6. gnConv proj_in (96 -> 192)
    gemm(stream, 0, 1, proj_in_w, bufA, proj_in_b, nullptr, bufF, 192, 96, 0,0,0,0, 96L*L, 192L*L, 0, 1.f);
    // 7. SS2D in_proj (168 -> 672), input = fused rows [24,192)
    gemm(stream, 0, 1, in_proj_w, bufF + 24*L, nullptr, nullptr, bufXZ, 672, 168, 0,0,0,0, 192L*L, 672L*L, 0, 1.f);
    // 8. depthwise 3x3 + SiLU
    dwconv_silu<<<dim3(16, 4*336), 256, 0, stream>>>(bufXZ, conv_w, conv_b, bufXI);
    // 9. x_proj per direction (43 x 336), natural pixel order
    for (int k = 0; k < 4; ++k)
        gemm(stream, 0, 1, x_proj_w + (long)k*43*336, bufXI, nullptr, nullptr,
             bufC + (long)k*43*L, 43, 336, 0,0,0,0, 336L*L, 4L*43*L, 0, 1.f);
    // 10. four-direction selective scan (dt_proj + softplus fused, fp32)
    scan_kernel<<<(4*4*336 + 255)/256, 256, 0, stream>>>(bufXI, bufC, dt_proj_w, dt_proj_b, A_log, Dp, bufOY);
    // 11. combine scattered direction planes
    combine_kernel<<<(int)((4L*336*L + 255)/256), 256, 0, stream>>>(bufOY, bufXI);
    // 12. out_norm LN (eps 1e-5) * silu(z)
    ln_gate_kernel<<<(4*LPIX+255)/256, 256, 0, stream>>>(bufXI, out_nw, out_nb, bufXZ, bufY2);
    // 13. out_proj (336 -> 168), fold HorNet 1/3 scale
    gemm(stream, 0, 1, out_proj_w, bufY2, nullptr, nullptr, bufDW, 168, 336, 0,0,0,0, 336L*L, 168L*L, 0, 1.f/3.f);
    // 14. gnConv: y = pwa * d0
    mul_kernel<<<(int)((4L*24*L + 255)/256), 256, 0, stream>>>(bufF, 192L*L, bufDW, 168L*L, bufGA, 24L*L, 24);
    // 15. (pw0 @ y + b) * d1
    gemm(stream, 0, 1, pw0_w, bufGA, pw0_b, bufDW + 24*L, bufGB, 48, 24, 0,0,0,0, 24L*L, 48L*L, 168L*L, 1.f);
    // 16. (pw1 @ y + b) * d2
    gemm(stream, 0, 1, pw1_w, bufGB, pw1_b, bufDW + 72*L, bufB, 96, 48, 0,0,0,0, 48L*L, 96L*L, 168L*L, 1.f);
    // 17. proj_out
    gemm(stream, 0, 1, proj_out_w, bufB, proj_out_b, nullptr, bufA, 96, 96, 0,0,0,0, 96L*L, 96L*L, 0, 1.f);
    // 18. residual add (msdac out + attn)
    add_kernel<<<(int)((4L*96*L + 255)/256), 256, 0, stream>>>(bufMY, bufA, bufB, 4L*96*L);
    // 19. final LN (1e-5) + GELU + skip*scale
    final_kernel<<<(4*LPIX+255)/256, 256, 0, stream>>>(bufB, ln_w, ln_b, bufS, scale, (float*)d_out);
}
